// GroupQueryattention_53171695125012
// MI455X (gfx1250) — matrix operand order repaired and checked by reading, not compiled, not run on hardware
//
#include <hip/hip_runtime.h>

#define HIDDEN 2048
#define HEAD_DIM 128
#define GROUPS 8
#define HEADS 16
#define NB 2
#define SL 2048
#define SKK 2048
#define KVW (HEAD_DIM * GROUPS) /* 1024 */

typedef __attribute__((ext_vector_type(16))) _Float16 v16h;
typedef __attribute__((ext_vector_type(8)))  _Float16 v8h;
typedef __attribute__((ext_vector_type(8)))  float    v8f;

union V16U { v16h v; v8h h[2]; };

static __device__ inline v8f wmma16(v16h a, v16h b, v8f c) {
  // D = A(16x32 f16) * B(32x16 f16) + C(f32)
  return __builtin_amdgcn_wmma_f32_16x16x32_f16(false, a, false, b, (short)0, c,
                                                false, false);
}

// Async Global -> LDS copy, 16 bytes per lane (ASYNCcnt-tracked, writes LDS
// directly, no VGPR staging).  lds_off is the LDS byte address (low 32 bits of
// a generic shared pointer), gaddr a per-lane 64-bit global address.
static __device__ inline void async_load_b128(unsigned int lds_off, const void* gaddr) {
  asm volatile("global_load_async_to_lds_b128 %0, %1, off"
               :: "v"(lds_off), "v"(gaddr) : "memory");
}
static __device__ inline void wait_asynccnt0() {
  asm volatile("s_wait_asynccnt 0x0" ::: "memory");
}

// A-operand from an f32 row (convert to f16 on the fly).
// Layout: lane row m; VGPR0-7 = K(kb+16*hi .. +15)
static __device__ inline v16h load_a_f32(const float* __restrict__ row, int kb, int hi) {
  v8f c0 = *(const v8f*)(row + kb + 16 * hi);
  v8f c1 = *(const v8f*)(row + kb + 16 * hi + 8);
  v16h a;
#pragma unroll
  for (int i = 0; i < 8; ++i) {
    a[i]     = (_Float16)c0[i];
    a[8 + i] = (_Float16)c1[i];
  }
  return a;
}

// A-operand from an f16 row-major row.
static __device__ inline v16h load_a_f16(const _Float16* __restrict__ row, int kb, int hi) {
  V16U u;
  u.h[0] = *(const v8h*)(row + kb + 16 * hi);
  u.h[1] = *(const v8h*)(row + kb + 16 * hi + 8);
  return u.v;
}

// ---------------------------------------------------------------------------
// Weight transpose + f16 convert:  W[K][N] (f32) -> WT[N][K] (f16)
// LDS-tiled 32x32: coalesced reads along N, coalesced writes along K,
// padded tile (33) to avoid bank conflicts, shift-only indexing.
// Grid: (N/32, K/32), block: 256 threads.
// ---------------------------------------------------------------------------
__global__ __launch_bounds__(256) void transpose_w_f16(
    const float* __restrict__ W, _Float16* __restrict__ WT, int K, int N) {
  __shared__ _Float16 tile[32][33];
  int tx = threadIdx.x & 31;
  int ty = threadIdx.x >> 5;           // 0..7
  int n0 = blockIdx.x << 5;
  int k0 = blockIdx.y << 5;
#pragma unroll
  for (int i = 0; i < 4; ++i) {
    int kr = ty + i * 8;
    tile[kr][tx] = (_Float16)W[(size_t)(k0 + kr) * N + n0 + tx];
  }
  __syncthreads();
#pragma unroll
  for (int i = 0; i < 4; ++i) {
    int nr = ty + i * 8;
    WT[(size_t)(n0 + nr) * K + k0 + tx] = tile[tx][nr];
  }
}

// ---------------------------------------------------------------------------
// Projection GEMM:  out = X(f32)[M,K] @ W[K,N] + bias,  via WT[N][K] (f16).
// Block = 8 waves sharing one 128-wide N panel; wave computes a 32x128 tile.
// The 128x32 f16 B panel (8KB) is double-buffered in LDS and filled with
// ASYNC global->LDS copies (global_load_async_to_lds_b128): no VGPR staging,
// latency hidden under the 16-WMMA chain, s_wait_asynccnt + one barrier per
// K-step.
// transposed==0 : out f16 [M,N]
// transposed==1 : out f16 [b][N][2048] (token innermost), producing V^T.
// ---------------------------------------------------------------------------
__global__ __launch_bounds__(256) void gemm_proj(
    const float* __restrict__ X, const _Float16* __restrict__ WT,
    const float* __restrict__ bias, _Float16* __restrict__ out,
    int M, int N, int K, int transposed) {
  __shared__ alignas(64) _Float16 Bpanel[2][128 * 32];  // 2 x 8KB

  int tid = threadIdx.x;
  int wave = (int)((blockIdx.x * blockDim.x + tid) >> 5);
  int lane = tid & 31;
  int hi = lane >> 4;
  int ln = lane & 15;
  int mtiles = M >> 5;                 // 32-row tiles; multiple of 8
  int mtile = wave % mtiles;
  int n0 = (wave / mtiles) << 7;       // same for all 8 waves of the block
  int m0 = mtile << 5;

  const float* arow0 = X + (size_t)(m0 + ln) * K;
  const float* arow1 = X + (size_t)(m0 + 16 + ln) * K;

  // This thread's slice of the B panel: row nl, 16 f16 starting at k=16*half.
  int nl = tid & 127;
  int half = tid >> 7;
  const _Float16* wrow = WT + (size_t)(n0 + nl) * K + 16 * half;
  // LDS byte addresses of this thread's 32B slice in each buffer.
  unsigned int slot0 = (unsigned int)(uintptr_t)&Bpanel[0][nl * 32 + 16 * half];
  unsigned int slot1 = (unsigned int)(uintptr_t)&Bpanel[1][nl * 32 + 16 * half];

  // Prologue: stage panel for kb=0 asynchronously.
  async_load_b128(slot0, wrow);
  async_load_b128(slot0 + 16, wrow + 8);

  v8f zero = {};
  v8f acc[2][8];
#pragma unroll
  for (int j = 0; j < 8; ++j) { acc[0][j] = zero; acc[1][j] = zero; }

  wait_asynccnt0();
  __syncthreads();

  for (int kb = 0; kb < K; kb += 32) {
    int p = (kb >> 5) & 1;
    bool hasnext = (kb + 32) < K;

    // Kick off async fill of the other buffer with the next panel.
    if (hasnext) {
      unsigned int dst = p ? slot0 : slot1;
      const _Float16* gsrc = wrow + kb + 32;
      async_load_b128(dst, gsrc);
      async_load_b128(dst + 16, gsrc + 8);
    }

    // A tiles (f32 -> f16 on the fly).
    v16h a0 = load_a_f32(arow0, kb, hi);
    v16h a1 = load_a_f32(arow1, kb, hi);

    // B operands from LDS panel.
    const _Float16* bp = &Bpanel[p][0];
    v16h bt[8];
#pragma unroll
    for (int j = 0; j < 8; ++j) {
      bt[j] = *(const v16h*)(bp + (j * 16 + ln) * 32 + 16 * hi);
    }
#pragma unroll
    for (int j = 0; j < 8; ++j) {
      acc[0][j] = wmma16(a0, bt[j], acc[0][j]);
      acc[1][j] = wmma16(a1, bt[j], acc[1][j]);
    }

    if (hasnext) wait_asynccnt0();
    __syncthreads();
  }

#pragma unroll
  for (int j = 0; j < 8; ++j) {
    int ncol = n0 + j * 16 + ln;
    float bv = bias[ncol];
#pragma unroll
    for (int mm = 0; mm < 2; ++mm) {
#pragma unroll
      for (int g = 0; g < 8; ++g) {
        int row = m0 + mm * 16 + g + 8 * hi;
        float val = acc[mm][j][g] + bv;
        if (!transposed) {
          out[(size_t)row * N + ncol] = (_Float16)val;
        } else {
          int bb = row >> 11;      // batch (seq len 2048)
          int tok = row & 2047;
          out[((size_t)bb * N + ncol) * 2048 + tok] = (_Float16)val;
        }
      }
    }
  }
}

// ---------------------------------------------------------------------------
// Flash GQA attention. One wave per (batch, head, 16-query tile).
// qp: f16 [NB][SL][HIDDEN]   kp: f16 [NB][SKK][KVW]   vT: f16 [NB][KVW][SKK]
// out: f32 [NB][SL][HIDDEN]
// ---------------------------------------------------------------------------
__global__ __launch_bounds__(128) void gqa_attn(
    const _Float16* __restrict__ qp, const _Float16* __restrict__ kp,
    const _Float16* __restrict__ vT, float* __restrict__ out) {
  __shared__ alignas(64) _Float16 lds[4][16 * 32];

  int wv = threadIdx.x >> 5;
  int lane = threadIdx.x & 31;
  int hi = lane >> 4;
  int ln = lane & 15;

  int gw = blockIdx.x * 4 + wv;
  int qtile = gw & 127;          // SL/16 = 128
  int h = (gw >> 7) & 15;        // head
  int b = gw >> 11;              // batch
  int g = h >> 1;                // KV group (HPG = 2)

  // Resident Q tile: 4 A-operands covering d = 0..127 (K=32 each).
  const _Float16* qrow = qp + ((size_t)b * SL + qtile * 16 + ln) * HIDDEN + h * HEAD_DIM;
  v16h Qa[4];
#pragma unroll
  for (int t = 0; t < 4; ++t) Qa[t] = load_a_f16(qrow, t * 32, hi);

  const _Float16* kbase = kp + (size_t)b * SKK * KVW + g * HEAD_DIM;
  const _Float16* vbase = vT + ((size_t)b * KVW + g * HEAD_DIM) * SKK;
  _Float16* myl = lds[wv];

  const float scale = 0.08838834764831845f; // 1/sqrt(128)
  float Mr[8], Lr[8];
  v8f zero = {};
  v8f acc[8];
#pragma unroll
  for (int r = 0; r < 8; ++r) { Mr[r] = -1e30f; Lr[r] = 0.0f; acc[r] = zero; }

  for (int kb = 0; kb < SKK; kb += 32) {
    // Prefetch next key block's K rows (gfx1250 global_prefetch path).
    if (kb + 32 < SKK) {
      __builtin_prefetch(kbase + (size_t)(kb + 32 + ln) * KVW, 0, 3);
      __builtin_prefetch(kbase + (size_t)(kb + 48 + ln) * KVW, 0, 3);
    }

    // ----- scores: S0 keys [kb..kb+15], S1 keys [kb+16..kb+31] -----
    const _Float16* k0 = kbase + (size_t)(kb + ln) * KVW;
    const _Float16* k1 = kbase + (size_t)(kb + 16 + ln) * KVW;
    v16h kt0[4], kt1[4];
#pragma unroll
    for (int t = 0; t < 4; ++t) {
      kt0[t] = *(const v16h*)(k0 + t * 32 + 16 * hi);
      kt1[t] = *(const v16h*)(k1 + t * 32 + 16 * hi);
    }
    v8f S0 = zero, S1 = zero;
#pragma unroll
    for (int t = 0; t < 4; ++t) {
      S0 = wmma16(Qa[t], kt0[t], S0);
      S1 = wmma16(Qa[t], kt1[t], S1);
    }

    // Issue V-tile loads now; their latency is hidden under the softmax
    // VALU/transcendental work below (no dependence on it).
    v16h vb[8];
#pragma unroll
    for (int j = 0; j < 8; ++j) {
      vb[j] = *(const v16h*)(vbase + (size_t)(j * 16 + ln) * SKK + kb + 16 * hi);
    }

    // ----- online softmax update; write exp'ed probs (f16) to LDS -----
    float alpha[8];
#pragma unroll
    for (int r = 0; r < 8; ++r) {
      float s0 = S0[r] * scale;
      float s1 = S1[r] * scale;
      float t = fmaxf(s0, s1);
      t = fmaxf(t, __shfl_xor(t, 1));
      t = fmaxf(t, __shfl_xor(t, 2));
      t = fmaxf(t, __shfl_xor(t, 4));
      t = fmaxf(t, __shfl_xor(t, 8));   // row max within 16-lane half
      float mn = fmaxf(Mr[r], t);
      alpha[r] = __expf(Mr[r] - mn);
      Mr[r] = mn;
      float p0 = __expf(s0 - mn);
      float p1 = __expf(s1 - mn);
      // C-layout row (r + 8*hi) -> LDS row-major 16x32 tile (A-layout source)
      myl[(r + 8 * hi) * 32 + ln]      = (_Float16)p0;
      myl[(r + 8 * hi) * 32 + 16 + ln] = (_Float16)p1;
      float rs = p0 + p1;
      rs += __shfl_xor(rs, 1);
      rs += __shfl_xor(rs, 2);
      rs += __shfl_xor(rs, 4);
      rs += __shfl_xor(rs, 8);
      Lr[r] = Lr[r] * alpha[r] + rs;
    }

    asm volatile("s_wait_dscnt 0" ::: "memory"); // CDNA5 split DS counter

    // ----- P(16x32) @ V(32x128), α-rescale accumulators -----
    v16h Pa = load_a_f16(myl + ln * 32, 0, hi);
#pragma unroll
    for (int j = 0; j < 8; ++j) {
#pragma unroll
      for (int r = 0; r < 8; ++r) acc[j][r] *= alpha[r];
    }
#pragma unroll
    for (int j = 0; j < 8; ++j) {
      acc[j] = wmma16(Pa, vb[j], acc[j]);
    }
  }

  // ----- epilogue: out = acc / L -----
  float* orow = out + ((size_t)b * SL + qtile * 16) * HIDDEN + h * HEAD_DIM;
#pragma unroll
  for (int j = 0; j < 8; ++j) {
#pragma unroll
    for (int r = 0; r < 8; ++r) {
      orow[(size_t)(r + 8 * hi) * HIDDEN + j * 16 + ln] = acc[j][r] / Lr[r];
    }
  }
}

// ---------------------------------------------------------------------------
extern "C" void kernel_launch(void* const* d_in, const int* in_sizes, int n_in,
                              void* d_out, int out_size, void* d_ws, size_t ws_size,
                              hipStream_t stream) {
  const float* q  = (const float*)d_in[0];
  const float* k  = (const float*)d_in[1];
  const float* v  = (const float*)d_in[2];
  const float* Wq = (const float*)d_in[3];
  const float* bq = (const float*)d_in[4];
  const float* Wk = (const float*)d_in[5];
  const float* bk = (const float*)d_in[6];
  const float* Wv = (const float*)d_in[7];
  const float* bv = (const float*)d_in[8];
  float* out = (float*)d_out;

  char* ws = (char*)d_ws;
  _Float16* WqT = (_Float16*)ws; ws += (size_t)HIDDEN * HIDDEN * sizeof(_Float16);
  _Float16* WkT = (_Float16*)ws; ws += (size_t)HIDDEN * KVW * sizeof(_Float16);
  _Float16* WvT = (_Float16*)ws; ws += (size_t)HIDDEN * KVW * sizeof(_Float16);
  _Float16* qpf = (_Float16*)ws; ws += (size_t)NB * SL * HIDDEN * sizeof(_Float16);
  _Float16* kpf = (_Float16*)ws; ws += (size_t)NB * SKK * KVW * sizeof(_Float16);
  _Float16* vTf = (_Float16*)ws; ws += (size_t)NB * KVW * SKK * sizeof(_Float16);
  (void)ws_size; (void)in_sizes; (void)n_in; (void)out_size;

  // 1) Transpose weights to K-major f16 (B-operand wants contiguous K).
  transpose_w_f16<<<dim3(HIDDEN / 32, HIDDEN / 32), 256, 0, stream>>>(Wq, WqT, HIDDEN, HIDDEN);
  transpose_w_f16<<<dim3(KVW / 32, HIDDEN / 32), 256, 0, stream>>>(Wk, WkT, HIDDEN, KVW);
  transpose_w_f16<<<dim3(KVW / 32, HIDDEN / 32), 256, 0, stream>>>(Wv, WvT, HIDDEN, KVW);

  // 2) Projections (WMMA). M = NB*seq = 4096, 32x128 tile per wave,
  //    8 waves/block share an async-LDS-staged B panel.
  {
    int M = NB * SL;
    int waves = (M / 32) * (HIDDEN / 128);   // 2048
    gemm_proj<<<waves / 8, 256, 0, stream>>>(q, WqT, bq, qpf, M, HIDDEN, HIDDEN, 0);
  }
  {
    int M = NB * SKK;
    int waves = (M / 32) * (KVW / 128);      // 1024
    gemm_proj<<<waves / 8, 256, 0, stream>>>(k, WkT, bk, kpf, M, KVW, HIDDEN, 0);
    gemm_proj<<<waves / 8, 256, 0, stream>>>(v, WvT, bv, vTf, M, KVW, HIDDEN, 1);
  }

  // 3) Flash GQA attention (WMMA): NB*HEADS*(SL/16) = 4096 waves, 4 per block.
  gqa_attn<<<(NB * HEADS * (SL / 16)) / 4, 128, 0, stream>>>(qpf, kpf, vTf, out);
}
